// SkipGPTMLP_8306466750646
// MI455X (gfx1250) — matrix operand order repaired and checked by reading, not compiled, not run on hardware
//
#include <hip/hip_runtime.h>
#include <hip/hip_bf16.h>

// ---------------------------------------------------------------------------
// SwiGLU MLP with per-token route mask, MI455X (gfx1250, wave32, WMMA).
//   pass1: h = silu(x@wg^T) * (x@wu^T) * mask   [4096,11008] bf16 (workspace)
//   pass2: out = h @ wd^T                        [4096,4096]  f32
// bf16 WMMA (v_wmma_f32_16x16x32_bf16), f32 accumulate, LDS double-buffered.
// fp32 operands are converted to bf16 on the fly while staging into LDS (no
// separate conversion pass: the harness times every replay, and a standalone
// convert pass would move more HBM bytes than both GEMMs).  L2 reuse comes
// from launch order: blockIdx.x walks M, so consecutive blocks share one
// small weight band, and x (64MB) lives in the 192MB L2 across the y sweep.
// Pass-2's bf16 A tiles are staged with global_load_async_to_lds_b128
// (ASYNCcnt) overlapping the WMMA chain.
// ---------------------------------------------------------------------------

typedef __bf16 bf16_t;
typedef bf16_t v16bf __attribute__((ext_vector_type(16)));
typedef float  v8f   __attribute__((ext_vector_type(8)));

#define MDIM 4096
#define HDIM 4096
#define IDIM 11008
#define LDS_STRIDE 80   // bytes per 32-element bf16 row (64B data + 16B pad)

static __device__ __forceinline__ unsigned pack_bf16x2(float a, float b) {
    union { bf16_t h[2]; unsigned u; } v;
    v.h[0] = (bf16_t)a;
    v.h[1] = (bf16_t)b;
    return v.u;
}

static __device__ __forceinline__ unsigned short f32_to_bf16(float f) {
    union { bf16_t h; unsigned short s; } v;
    v.h = (bf16_t)f;
    return v.s;
}

union FragBF { v16bf v; uint4 q[2]; };

// A operand: 16x32 bf16 tile (lane = M row; K chunks 0-7/16-23 or 8-15/24-31)
static __device__ __forceinline__ v16bf ldsFragA(const unsigned char* sm, int rowBase, int lane) {
    int r = rowBase + (lane & 15);
    int kb = (lane & 16) ? 16 : 0;
    FragBF f;
    f.q[0] = *(const uint4*)(sm + r * LDS_STRIDE + kb);
    f.q[1] = *(const uint4*)(sm + r * LDS_STRIDE + kb + 32);
    return f.v;
}

// B operand: 32x16 bf16 (lane = N col; K chunks 0-7/16-23 or 8-15/24-31)
static __device__ __forceinline__ v16bf ldsFragB(const unsigned char* sm, int rowBase, int lane) {
    int r = rowBase + (lane & 15);
    int kb = (lane & 16) ? 16 : 0;
    FragBF f;
    f.q[0] = *(const uint4*)(sm + r * LDS_STRIDE + kb);
    f.q[1] = *(const uint4*)(sm + r * LDS_STRIDE + kb + 32);
    return f.v;
}

// Stage a (rows x 32) fp32 tile into LDS as bf16. One unit = 8 elements.
static __device__ __forceinline__ void stageTileF32(unsigned char* sm, const float* src,
                                                    size_t srcStride, int unit) {
    int row = unit >> 2, chunk = unit & 3;
    const float* s = src + (size_t)row * srcStride + chunk * 8;
    float4 a = *(const float4*)(s);
    float4 b = *(const float4*)(s + 4);
    uint4 q;
    q.x = pack_bf16x2(a.x, a.y);
    q.y = pack_bf16x2(a.z, a.w);
    q.z = pack_bf16x2(b.x, b.y);
    q.w = pack_bf16x2(b.z, b.w);
    *(uint4*)(sm + row * LDS_STRIDE + chunk * 16) = q;
}

// Stage a (rows x 32) bf16 tile straight into LDS with the async data mover
// (global_load_async_to_lds_b128, tracked by ASYNCcnt; no VGPR data path).
static __device__ __forceinline__ void stageAsyncBF16(unsigned char* sm, const unsigned short* src,
                                                      size_t srcStride, int unit) {
    int row = unit >> 2, chunk = unit & 3;
    const unsigned short* g = src + (size_t)row * srcStride + chunk * 8;
    unsigned lds = (unsigned)(unsigned long long)(sm + row * LDS_STRIDE + chunk * 16);
    asm volatile("global_load_async_to_lds_b128 %0, %1, off"
                 :: "v"(lds), "v"(g) : "memory");
}

#define WAIT_ASYNC() asm volatile("s_wait_asynccnt 0x0" ::: "memory")

#define WMMA_BF16(A, B, C) \
    __builtin_amdgcn_wmma_f32_16x16x32_bf16(false, (A), false, (B), (short)0, (C), false, false)

#define V8F_ZERO {0,0,0,0,0,0,0,0}

// ---------------------------------------------------------------------------
// Pass 1: h[m,i] = silu(x@wg^T) * (x@wu^T) * mask[m], bf16 out.
// Block 128(M) x 64(I). 8 waves: (wm 0..3 x 32 rows) x (wn 0..1 x 32 cols).
// Per wave / K-step: 2 A frags, 2 Bu + 2 Bg frags, 8 WMMAs. Double-buffered.
// ---------------------------------------------------------------------------
__global__ __launch_bounds__(256) void k_up_gate(const float* __restrict__ x,
                                                 const float* __restrict__ wu,
                                                 const float* __restrict__ wg,
                                                 const int* __restrict__ route,
                                                 unsigned short* __restrict__ hmid) {
    __shared__ unsigned char smA[2 * 128 * LDS_STRIDE];
    __shared__ unsigned char smU[2 *  64 * LDS_STRIDE];
    __shared__ unsigned char smG[2 *  64 * LDS_STRIDE];
    const int SZA = 128 * LDS_STRIDE, SZB = 64 * LDS_STRIDE;

    const int tid  = threadIdx.x;
    const int lane = tid & 31;
    const int wave = tid >> 5;
    const int wm   = (wave & 3) * 32;
    const int wn   = (wave >> 2) * 32;
    const int blkM = blockIdx.x * 128;   // x fastest: M-blocks share the weight band in L2
    const int blkN = blockIdx.y * 64;

    const float* xA  = x  + (size_t)blkM * HDIM;
    const float* wuB = wu + (size_t)blkN * HDIM;
    const float* wgB = wg + (size_t)blkN * HDIM;

    v8f u00 = V8F_ZERO, u01 = V8F_ZERO, u10 = V8F_ZERO, u11 = V8F_ZERO;
    v8f g00 = V8F_ZERO, g01 = V8F_ZERO, g10 = V8F_ZERO, g11 = V8F_ZERO;

    // prologue: stage tile 0 into buffer 0
    stageTileF32(smA, xA, HDIM, tid);
    stageTileF32(smA, xA, HDIM, tid + 256);
    stageTileF32(smU, wuB, HDIM, tid);
    stageTileF32(smG, wgB, HDIM, tid);

    const int KT = HDIM / 32;
    for (int kt = 0; kt < KT; ++kt) {
        __syncthreads();  // cur buffer writes visible; prev reads of other buffer done
        const int cur = kt & 1;
        unsigned char* cA = smA + cur * SZA;
        unsigned char* cU = smU + cur * SZB;
        unsigned char* cG = smG + cur * SZB;
        if (kt + 1 < KT) {
            const int k1 = (kt + 1) * 32;
            unsigned char* nA = smA + (cur ^ 1) * SZA;
            unsigned char* nU = smU + (cur ^ 1) * SZB;
            unsigned char* nG = smG + (cur ^ 1) * SZB;
            stageTileF32(nA, xA  + k1, HDIM, tid);
            stageTileF32(nA, xA  + k1, HDIM, tid + 256);
            stageTileF32(nU, wuB + k1, HDIM, tid);
            stageTileF32(nG, wgB + k1, HDIM, tid);
        }
        v16bf a0  = ldsFragA(cA, wm,      lane);
        v16bf a1  = ldsFragA(cA, wm + 16, lane);
        v16bf bu0 = ldsFragB(cU, wn,      lane);
        v16bf bu1 = ldsFragB(cU, wn + 16, lane);
        v16bf bg0 = ldsFragB(cG, wn,      lane);
        v16bf bg1 = ldsFragB(cG, wn + 16, lane);
        u00 = WMMA_BF16(a0, bu0, u00);
        u01 = WMMA_BF16(a0, bu1, u01);
        u10 = WMMA_BF16(a1, bu0, u10);
        u11 = WMMA_BF16(a1, bu1, u11);
        g00 = WMMA_BF16(a0, bg0, g00);
        g01 = WMMA_BF16(a0, bg1, g01);
        g10 = WMMA_BF16(a1, bg0, g10);
        g11 = WMMA_BF16(a1, bg1, g11);
    }

    // epilogue: SiLU(gate) * up * route-mask -> bf16
    const int rbase = blkM + wm + ((lane & 16) ? 8 : 0);
    const int cbase = blkN + wn + (lane & 15);
#pragma unroll
    for (int i = 0; i < 2; ++i) {
        const v8f* U0 = i ? &u10 : &u00;
        const v8f* U1 = i ? &u11 : &u01;
        const v8f* G0 = i ? &g10 : &g00;
        const v8f* G1 = i ? &g11 : &g01;
#pragma unroll
        for (int r = 0; r < 8; ++r) {
            int m = rbase + i * 16 + r;
            float mk = (float)route[m];
            float gg0 = (*G0)[r], gg1 = (*G1)[r];
            float h0 = (*U0)[r] * (gg0 / (1.0f + __expf(-gg0))) * mk;
            float h1 = (*U1)[r] * (gg1 / (1.0f + __expf(-gg1))) * mk;
            hmid[(size_t)m * IDIM + cbase]      = f32_to_bf16(h0);
            hmid[(size_t)m * IDIM + cbase + 16] = f32_to_bf16(h1);
        }
    }
}

// ---------------------------------------------------------------------------
// Pass 2: out[m,h] = hmid @ w_down^T, fp32 out.
// Block 128(M) x 128(H). 8 waves: (wm 0..3 x 32 rows) x (wn 0..1 x 64 cols).
// A (bf16 intermediate) staged via global_load_async_to_lds_b128;
// B (fp32 weights) staged via load+cvt+ds_store. Double-buffered.
// ---------------------------------------------------------------------------
__global__ __launch_bounds__(256) void k_down(const unsigned short* __restrict__ h,
                                              const float* __restrict__ wd,
                                              float* __restrict__ out) {
    __shared__ unsigned char smA[2 * 128 * LDS_STRIDE];
    __shared__ unsigned char smB[2 * 128 * LDS_STRIDE];
    const int SZ = 128 * LDS_STRIDE;

    const int tid  = threadIdx.x;
    const int lane = tid & 31;
    const int wave = tid >> 5;
    const int wm   = (wave & 3) * 32;
    const int wn   = (wave >> 2) * 64;
    const int blkM = blockIdx.x * 128;   // x fastest: M-blocks share the weight band in L2
    const int blkN = blockIdx.y * 128;

    const unsigned short* hA = h  + (size_t)blkM * IDIM;
    const float*          wB = wd + (size_t)blkN * IDIM;

    v8f c00 = V8F_ZERO, c01 = V8F_ZERO, c02 = V8F_ZERO, c03 = V8F_ZERO;
    v8f c10 = V8F_ZERO, c11 = V8F_ZERO, c12 = V8F_ZERO, c13 = V8F_ZERO;

    stageAsyncBF16(smA, hA, IDIM, tid);
    stageAsyncBF16(smA, hA, IDIM, tid + 256);
    stageTileF32(smB, wB, IDIM, tid);
    stageTileF32(smB, wB, IDIM, tid + 256);

    const int KT = IDIM / 32;
    for (int kt = 0; kt < KT; ++kt) {
        WAIT_ASYNC();      // async fills of the buffer we are about to read
        __syncthreads();
        const int cur = kt & 1;
        unsigned char* cA = smA + cur * SZ;
        unsigned char* cB = smB + cur * SZ;
        if (kt + 1 < KT) {
            const int k1 = (kt + 1) * 32;
            unsigned char* nA = smA + (cur ^ 1) * SZ;
            unsigned char* nB = smB + (cur ^ 1) * SZ;
            stageAsyncBF16(nA, hA + k1, IDIM, tid);
            stageAsyncBF16(nA, hA + k1, IDIM, tid + 256);
            stageTileF32(nB, wB + k1, IDIM, tid);
            stageTileF32(nB, wB + k1, IDIM, tid + 256);
        }
        v16bf a0 = ldsFragA(cA, wm,      lane);
        v16bf a1 = ldsFragA(cA, wm + 16, lane);
        v16bf b0 = ldsFragB(cB, wn,      lane);
        v16bf b1 = ldsFragB(cB, wn + 16, lane);
        v16bf b2 = ldsFragB(cB, wn + 32, lane);
        v16bf b3 = ldsFragB(cB, wn + 48, lane);
        c00 = WMMA_BF16(a0, b0, c00);
        c01 = WMMA_BF16(a0, b1, c01);
        c02 = WMMA_BF16(a0, b2, c02);
        c03 = WMMA_BF16(a0, b3, c03);
        c10 = WMMA_BF16(a1, b0, c10);
        c11 = WMMA_BF16(a1, b1, c11);
        c12 = WMMA_BF16(a1, b2, c12);
        c13 = WMMA_BF16(a1, b3, c13);
    }

    const int rbase = blkM + wm + ((lane & 16) ? 8 : 0);
    const int cbase = blkN + wn + (lane & 15);
#pragma unroll
    for (int r = 0; r < 8; ++r) {
        size_t row0 = (size_t)(rbase + r) * HDIM;
        size_t row1 = (size_t)(rbase + 16 + r) * HDIM;
        out[row0 + cbase]      = c00[r];
        out[row0 + cbase + 16] = c01[r];
        out[row0 + cbase + 32] = c02[r];
        out[row0 + cbase + 48] = c03[r];
        out[row1 + cbase]      = c10[r];
        out[row1 + cbase + 16] = c11[r];
        out[row1 + cbase + 32] = c12[r];
        out[row1 + cbase + 48] = c13[r];
    }
}

// ---------------------------------------------------------------------------
extern "C" void kernel_launch(void* const* d_in, const int* in_sizes, int n_in,
                              void* d_out, int out_size, void* d_ws, size_t ws_size,
                              hipStream_t stream) {
    const float* x      = (const float*)d_in[0];
    const float* w_up   = (const float*)d_in[1];
    const float* w_gate = (const float*)d_in[2];
    const float* w_down = (const float*)d_in[3];
    const int*   route  = (const int*)d_in[4];
    float*       out    = (float*)d_out;

    unsigned short* hm = (unsigned short*)d_ws;   // [4096, 11008] bf16 (90.2 MB)

    k_up_gate<<<dim3(MDIM / 128, IDIM / 64), 256, 0, stream>>>(x, w_up, w_gate, route, hm);
    k_down<<<dim3(MDIM / 128, HDIM / 128), 256, 0, stream>>>(hm, w_down, out);
}
